// GNNModel0_48241072668818
// MI455X (gfx1250) — compile-verified, matrix operand order checked by reading, not run on hardware
//
#include <hip/hip_runtime.h>
#include <hip/hip_bf16.h>

typedef __attribute__((ext_vector_type(16))) _Float16 v16h;
typedef __attribute__((ext_vector_type(8)))  float    v8f;

#define N_NODES   50000
#define N_EDGES   800000
#define N_GRAPHS  64
#define HID       128
#define N_CLASSES 38
#define EPS_BN    1e-5f

// ---- gfx1250 async global->LDS copy (ASYNCcnt path), per CDNA5 ISA ----
__device__ __forceinline__ void async_copy_b128(unsigned lds_addr, const void* gaddr) {
  // GV mode: per-lane 16B transfer, LDS byte address in VDST vgpr, 64-bit global addr in vaddr pair.
  asm volatile("global_load_async_to_lds_b128 %0, %1, off"
               :: "v"(lds_addr), "v"(gaddr)
               : "memory");
}
__device__ __forceinline__ void async_wait_all() {
  asm volatile("s_wait_asynccnt 0" ::: "memory");
}

// ---------------- generic zero ----------------
__global__ void zero_kernel(float* __restrict__ p, int n) {
  int i = blockIdx.x * blockDim.x + threadIdx.x;
  if (i < n) p[i] = 0.0f;
}

// ---------------- f32 -> f16 convert ----------------
__global__ void convert_f16_kernel(const float* __restrict__ in, _Float16* __restrict__ out, int n) {
  int i = blockIdx.x * blockDim.x + threadIdx.x;
  if (i < n) out[i] = (_Float16)in[i];
}

// ---------------- degree / dinv ----------------
__global__ void deg_kernel(const int* __restrict__ dst, const float* __restrict__ ew,
                           float* __restrict__ deg, int E) {
  int e = blockIdx.x * blockDim.x + threadIdx.x;
  if (e < E) atomicAdd(&deg[dst[e]], ew[e]);
}

__global__ void dinv_kernel(float* __restrict__ deg, int n) {
  int i = blockIdx.x * blockDim.x + threadIdx.x;
  if (i < n) {
    float d = deg[i] + 1.0f;              // self-loop weight 1
    d = fmaxf(d, 1e-12f);
    deg[i] = rsqrtf(d);
  }
}

// ---------------- WMMA GEMM: H[n x 128] = X16[n x 128] @ W16[128 x 128] (+bias) ----------------
// 256 threads = 8 waves. Block covers 64 rows; wave w owns column tile w (16 cols).
// X/W already f16: staged into LDS with gfx1250 async-to-LDS b128 copies; f32 accumulate
// via v_wmma_f32_16x16x32_f16.
__global__ __launch_bounds__(256)
void gemm128_wmma(const _Float16* __restrict__ X, const _Float16* __restrict__ W,
                  const float* __restrict__ bias, float* __restrict__ H, int n_rows) {
  __shared__ _Float16 sX[64 * 128];    // 16 KB
  __shared__ _Float16 sW[128 * 128];   // 32 KB
  const int tid  = threadIdx.x;
  const int wave = tid >> 5;
  const int lane = tid & 31;
  const int g    = lane >> 4;          // half-wave group
  const int ml   = lane & 15;
  const int row0 = blockIdx.x * 64;

  const unsigned ldsX0 = (unsigned)(size_t)&sX[0];
  const unsigned ldsW0 = (unsigned)(size_t)&sW[0];

  if (row0 + 64 <= n_rows) {
    // Full tile: contiguous 16 KB region -> pure async copy (4 x b128 per thread).
    const char* gX = (const char*)(X + (size_t)row0 * 128);
#pragma unroll
    for (int j = 0; j < 4; ++j) {
      unsigned o = (unsigned)(tid + j * 256) * 16u;
      async_copy_b128(ldsX0 + o, gX + o);
    }
  } else {
    // Tail block: scalar staging with zero padding.
    for (int idx = tid; idx < 64 * 128; idx += 256) {
      int r = idx >> 7, c = idx & 127;
      int gr = row0 + r;
      sX[idx] = (gr < n_rows) ? X[(size_t)gr * 128 + c] : (_Float16)0.0f;
    }
  }
  {
    // W: contiguous 32 KB -> async copy (8 x b128 per thread).
    const char* gW = (const char*)W;
#pragma unroll
    for (int j = 0; j < 8; ++j) {
      unsigned o = (unsigned)(tid + j * 256) * 16u;
      async_copy_b128(ldsW0 + o, gW + o);
    }
  }
  async_wait_all();
  __syncthreads();

  const int nt  = wave;                // column tile 0..7
  const int col = nt * 16 + ml;

  // B fragments (K blocks of 32), reused across all 4 M tiles.
  v16h bfrag[4];
#pragma unroll
  for (int kb = 0; kb < 4; ++kb) {
#pragma unroll
    for (int v = 0; v < 8; ++v) {
      int koff = ((v < 4) ? (2 * v) : (16 + 2 * (v - 4))) + 8 * g;
      int k = kb * 32 + koff;
      bfrag[kb][2 * v]     = sW[k * 128 + col];
      bfrag[kb][2 * v + 1] = sW[(k + 1) * 128 + col];
    }
  }

  const float bcol = (bias != nullptr) ? bias[col] : 0.0f;

#pragma unroll
  for (int mt = 0; mt < 4; ++mt) {
    v8f acc = {};
    const int arow = mt * 16 + ml;
#pragma unroll
    for (int kb = 0; kb < 4; ++kb) {
      v16h afrag;
#pragma unroll
      for (int v = 0; v < 8; ++v) {
        int koff = ((v < 4) ? (2 * v) : (16 + 2 * (v - 4))) + 8 * g;
        int k = kb * 32 + koff;
        afrag[2 * v]     = sX[arow * 128 + k];
        afrag[2 * v + 1] = sX[arow * 128 + k + 1];
      }
      acc = __builtin_amdgcn_wmma_f32_16x16x32_f16(false, afrag, false, bfrag[kb],
                                                   (short)0, acc, false, false);
    }
    const int rbase = row0 + mt * 16;
    if (rbase + 16 <= n_rows) {
      // Uniform fast path: straight-line stores, C layout M = r + 8*(lane>>4).
      float* out = H + (size_t)(rbase + 8 * g) * 128 + col;
#pragma unroll
      for (int r = 0; r < 8; ++r) out[(size_t)r * 128] = acc[r] + bcol;
    } else {
#pragma unroll
      for (int r = 0; r < 8; ++r) {
        int grow = rbase + r + 8 * g;
        if (grow < n_rows) H[(size_t)grow * 128 + col] = acc[r] + bcol;
      }
    }
  }
}

// ---------------- edge scatter: AGG[d] += H[s] * (dinv[s]*ew*dinv[d]) ----------------
// One wave per edge, each lane moves 4 features (float4 gather, 4 f32 atomics).
__global__ __launch_bounds__(256)
void scatter_kernel(const float* __restrict__ H, const int* __restrict__ src,
                    const int* __restrict__ dst, const float* __restrict__ ew,
                    const float* __restrict__ dinv, float* __restrict__ AGG, int E) {
  int wid  = (blockIdx.x * blockDim.x + threadIdx.x) >> 5;
  int lane = threadIdx.x & 31;
  if (wid >= E) return;
  int s = src[wid], d = dst[wid];
  float c = dinv[s] * ew[wid] * dinv[d];
  const float4* hp = (const float4*)(H + (size_t)s * 128);
  float4 h4 = hp[lane];
  float* out = AGG + (size_t)d * 128 + lane * 4;
  atomicAdd(out + 0, h4.x * c);
  atomicAdd(out + 1, h4.y * c);
  atomicAdd(out + 2, h4.z * c);
  atomicAdd(out + 3, h4.w * c);
}

// ---------------- self-loop + bias: AGG[i] += H[i]*dinv[i]^2 + b[f] ----------------
__global__ void selfloop_bias_kernel(const float* __restrict__ H, const float* __restrict__ dinv,
                                     const float* __restrict__ b, float* __restrict__ AGG, int n) {
  int i = blockIdx.x * blockDim.x + threadIdx.x;
  if (i >= n * HID) return;
  int node = i >> 7, f = i & 127;
  float di = dinv[node];
  AGG[i] += H[i] * di * di + b[f];
}

// ---------------- BN stats (sum / sumsq per feature) ----------------
__global__ __launch_bounds__(256)
void bn_stats_kernel(const float* __restrict__ X, float* __restrict__ stats, int n) {
  const int ROWS = 512;
  int f   = threadIdx.x & 127;
  int sub = threadIdx.x >> 7;                  // 0 or 1
  int r0  = blockIdx.x * ROWS + sub;
  int rend = blockIdx.x * ROWS + ROWS;
  if (rend > n) rend = n;
  float s = 0.f, ss = 0.f;
  for (int r = r0; r < rend; r += 2) {
    float v = X[(size_t)r * 128 + f];
    s += v; ss += v * v;
  }
  atomicAdd(&stats[f], s);
  atomicAdd(&stats[128 + f], ss);
}

__global__ void bn_finalize_kernel(const float* __restrict__ g, const float* __restrict__ bt,
                                   float* __restrict__ stats, int n) {
  int f = threadIdx.x;
  if (f < 128) {
    float inv_n = 1.0f / (float)n;
    float mean = stats[f] * inv_n;
    float var  = stats[128 + f] * inv_n - mean * mean;
    float sc   = g[f] * rsqrtf(var + EPS_BN);
    stats[256 + f] = sc;
    stats[384 + f] = bt[f] - mean * sc;
  }
}

// BN+ReLU; writes f32 (for BN-input / pooling consumers) and f16 (next GEMM's activations).
__global__ void bn_apply_relu_kernel(float* __restrict__ X, _Float16* __restrict__ out16,
                                     const float* __restrict__ stats, int n) {
  int i = blockIdx.x * blockDim.x + threadIdx.x;
  if (i >= n * HID) return;
  int f = i & 127;
  float v = X[i] * stats[256 + f] + stats[384 + f];
  v = v > 0.f ? v : 0.f;
  X[i] = v;
  out16[i] = (_Float16)v;
}

// ---------------- global add pool ----------------
__global__ void pool_kernel(const float* __restrict__ X, const int* __restrict__ batch,
                            float* __restrict__ pooled, int n) {
  int i = blockIdx.x * blockDim.x + threadIdx.x;
  if (i >= n * HID) return;
  int node = i >> 7, f = i & 127;
  atomicAdd(&pooled[batch[node] * HID + f], X[i]);
}

// ---------------- final classifier + log_softmax ----------------
__global__ __launch_bounds__(64)
void head_softmax_kernel(const float* __restrict__ t, const float* __restrict__ Wf,
                         const float* __restrict__ bf, float* __restrict__ out) {
  __shared__ float row[128];
  __shared__ float logits[N_CLASSES];
  __shared__ float red[2];
  int g = blockIdx.x;
  int tid = threadIdx.x;
  row[tid]      = t[g * 128 + tid];
  row[tid + 64] = t[g * 128 + tid + 64];
  __syncthreads();
  if (tid < N_CLASSES) {
    float acc = bf[tid];
    for (int k = 0; k < 128; ++k) acc += row[k] * Wf[k * N_CLASSES + tid];
    logits[tid] = acc;
  }
  __syncthreads();
  if (tid == 0) {
    float m = logits[0];
    for (int c = 1; c < N_CLASSES; ++c) m = fmaxf(m, logits[c]);
    float sum = 0.f;
    for (int c = 0; c < N_CLASSES; ++c) sum += __expf(logits[c] - m);
    red[0] = m;
    red[1] = __logf(sum);
  }
  __syncthreads();
  if (tid < N_CLASSES) out[g * N_CLASSES + tid] = logits[tid] - red[0] - red[1];
}

extern "C" void kernel_launch(void* const* d_in, const int* in_sizes, int n_in,
                              void* d_out, int out_size, void* d_ws, size_t ws_size,
                              hipStream_t stream) {
  (void)in_sizes; (void)n_in; (void)out_size; (void)ws_size;

  const float* x     = (const float*)d_in[0];
  const int*   ei    = (const int*)d_in[1];     // [2, E]
  const float* ew    = (const float*)d_in[2];
  const int*   batch = (const int*)d_in[3];
  const float* W1 = (const float*)d_in[4];  const float* b1 = (const float*)d_in[5];
  const float* g1 = (const float*)d_in[6];  const float* bt1 = (const float*)d_in[7];
  const float* W2 = (const float*)d_in[8];  const float* b2 = (const float*)d_in[9];
  const float* g2 = (const float*)d_in[10]; const float* bt2 = (const float*)d_in[11];
  const float* W3 = (const float*)d_in[12]; const float* b3 = (const float*)d_in[13];
  const float* Wn = (const float*)d_in[14]; const float* bn = (const float*)d_in[15];
  const float* Wf = (const float*)d_in[16]; const float* bf = (const float*)d_in[17];

  const int* src = ei;
  const int* dst = ei + N_EDGES;

  // workspace layout (16B-aligned sections): ~64.5 MB total
  _Float16* X16 = (_Float16*)d_ws;                       // N*HID halves (12.8 MB)
  _Float16* W16 = X16 + (size_t)N_NODES * HID;           // 4 * 128*128 halves (128 KB)
  _Float16* P16 = W16 + 4 * (HID * HID);                 // 64*128 halves (16 KB)
  float* B0     = (float*)(P16 + N_GRAPHS * HID);        // N*HID f32 (GEMM output H)
  float* B1     = B0 + (size_t)N_NODES * HID;            // N*HID f32 (AGG / activations)
  float* dinv   = B1 + (size_t)N_NODES * HID;            // N
  float* stats  = dinv + N_NODES;                        // 512 (sum, sumsq, scale, shift)
  float* pooled = stats + 512;                           // 64*128
  float* thead  = pooled + N_GRAPHS * HID;               // 64*128

  const int NT = 256;
  dim3 blk(NT);
  const int NF             = N_NODES * HID;
  const int nf_blocks      = (NF + NT - 1) / NT;
  const int gemm_blocks    = (N_NODES + 63) / 64;
  const int scatter_blocks = (N_EDGES * 32 + NT - 1) / NT;   // one wave per edge
  const int stats_blocks   = (N_NODES + 511) / 512;
  const int w_blocks       = (HID * HID + NT - 1) / NT;

  // One-time per launch: f16 copies of the 128x128 weights and the input activations.
  convert_f16_kernel<<<w_blocks, blk, 0, stream>>>(W1, W16 + 0 * HID * HID, HID * HID);
  convert_f16_kernel<<<w_blocks, blk, 0, stream>>>(W2, W16 + 1 * HID * HID, HID * HID);
  convert_f16_kernel<<<w_blocks, blk, 0, stream>>>(W3, W16 + 2 * HID * HID, HID * HID);
  convert_f16_kernel<<<w_blocks, blk, 0, stream>>>(Wn, W16 + 3 * HID * HID, HID * HID);
  convert_f16_kernel<<<nf_blocks, blk, 0, stream>>>(x, X16, NF);

  // graph normalization: deg -> dinv (fixed across layers, computed once)
  zero_kernel<<<(N_NODES + NT - 1) / NT, blk, 0, stream>>>(dinv, N_NODES);
  deg_kernel<<<(N_EDGES + NT - 1) / NT, blk, 0, stream>>>(dst, ew, dinv, N_EDGES);
  dinv_kernel<<<(N_NODES + NT - 1) / NT, blk, 0, stream>>>(dinv, N_NODES);

  struct Layer { const _Float16* W; const float* b; const float* g; const float* bt; };
  // NOTE: layer 3 deliberately uses g2/bt2 (reference reproduces that model bug).
  Layer layers[3] = { {W16 + 0 * HID * HID, b1, g1, bt1},
                      {W16 + 1 * HID * HID, b2, g2, bt2},
                      {W16 + 2 * HID * HID, b3, g2, bt2} };

  for (int l = 0; l < 3; ++l) {
    gemm128_wmma<<<gemm_blocks, blk, 0, stream>>>(X16, layers[l].W, nullptr, B0, N_NODES);
    zero_kernel<<<nf_blocks, blk, 0, stream>>>(B1, NF);
    scatter_kernel<<<scatter_blocks, blk, 0, stream>>>(B0, src, dst, ew, dinv, B1, N_EDGES);
    selfloop_bias_kernel<<<nf_blocks, blk, 0, stream>>>(B0, dinv, layers[l].b, B1, N_NODES);
    zero_kernel<<<(512 + NT - 1) / NT, blk, 0, stream>>>(stats, 512);
    bn_stats_kernel<<<stats_blocks, blk, 0, stream>>>(B1, stats, N_NODES);
    bn_finalize_kernel<<<1, 128, 0, stream>>>(layers[l].g, layers[l].bt, stats, N_NODES);
    bn_apply_relu_kernel<<<nf_blocks, blk, 0, stream>>>(B1, X16, stats, N_NODES);
  }

  // global add pool + head (pooled @ Wn + bn via the same WMMA kernel)
  zero_kernel<<<(N_GRAPHS * HID + NT - 1) / NT, blk, 0, stream>>>(pooled, N_GRAPHS * HID);
  pool_kernel<<<nf_blocks, blk, 0, stream>>>(B1, batch, pooled, N_NODES);
  convert_f16_kernel<<<(N_GRAPHS * HID + NT - 1) / NT, blk, 0, stream>>>(pooled, P16, N_GRAPHS * HID);
  gemm128_wmma<<<(N_GRAPHS + 63) / 64, blk, 0, stream>>>(P16, W16 + 3 * HID * HID, bn, thead, N_GRAPHS);
  head_softmax_kernel<<<N_GRAPHS, 64, 0, stream>>>(thead, Wf, bf, (float*)d_out);
}
